// VQEncoder_7447473291876
// MI455X (gfx1250) — compile-verified, matrix operand order checked by reading, not run on hardware
//
#include <hip/hip_runtime.h>
#include <math.h>
#include <stdint.h>

// ---------------- model dims (compile-time) ----------------
#define B_   16
#define NP_  1000
#define D_   256
#define H_   8
#define DH_  32
#define FF_  1024
#define L_   4
#define K_   1024
#define CP_  512
#define M_   (B_*NP_)      // 16000 rows through every GEMM
#define EPS_ 1e-5f

typedef __attribute__((ext_vector_type(16))) _Float16 v16h;
typedef __attribute__((ext_vector_type(8)))  _Float16 h8;
typedef __attribute__((ext_vector_type(8)))  float    v8f;

// ---------------- WMMA helpers (CDNA5 16x16x32 f16 -> f32) ----------------
__device__ __forceinline__ v8f wmma_16x16x32(v16h a, v16h b, v8f c) {
  return __builtin_amdgcn_wmma_f32_16x16x32_f16(false, a, false, b, (short)0, c,
                                                false, false);
}

// Async global->LDS copy of 16 bytes (gfx1250, ASYNCcnt-tracked).
__device__ __forceinline__ void async_copy16(void* lds_dst, const void* gsrc) {
  const uint32_t loff = (uint32_t)(uintptr_t)lds_dst;   // LDS aperture: low 32 bits
  asm volatile("global_load_async_to_lds_b128 %0, %1, off"
               :: "v"(loff), "v"((uint64_t)(uintptr_t)gsrc)
               : "memory");
}
__device__ __forceinline__ void wait_async0() {
  asm volatile("s_wait_asynccnt 0" ::: "memory");
}
__device__ __forceinline__ void wait_async2() {
  asm volatile("s_wait_asynccnt 2" ::: "memory");
}
__device__ __forceinline__ void wait_async3() {
  asm volatile("s_wait_asynccnt 3" ::: "memory");
}

// Read a 16x32 tile stored row-major in LDS (stride ld f16) into an A-frag
// (or B-frag when source holds B^T). ISA 7.12.2 layout.
__device__ __forceinline__ v16h frag_rowmajor(const _Float16* __restrict__ src, int ld) {
  const int lane = threadIdx.x & 31;
  const int r = lane & 15, half = lane >> 4;
  v16h f;
#pragma unroll
  for (int i = 0; i < 8; ++i) {
    const int k0 = ((i & 3) << 1) + (half << 3) + ((i >> 2) << 4);
    f[2*i]   = src[r*ld + k0];
    f[2*i+1] = src[r*ld + k0 + 1];
  }
  return f;
}

// B-fragment from k-major storage: B[k][n] = src[k*ld + n]  (used for P@V).
__device__ __forceinline__ v16h frag_colmajor(const _Float16* __restrict__ src, int ld) {
  const int lane = threadIdx.x & 31;
  const int n = lane & 15, half = lane >> 4;
  v16h f;
#pragma unroll
  for (int i = 0; i < 8; ++i) {
    const int k0 = ((i & 3) << 1) + (half << 3) + ((i >> 2) << 4);
    f[2*i]   = src[k0*ld + n];
    f[2*i+1] = src[(k0+1)*ld + n];
  }
  return f;
}

__device__ __forceinline__ float gelu_exact(float x) {
  return 0.5f * x * (1.0f + erff(x * 0.70710678118654752f));
}

// ---------------- weight convert f32 -> f16 ----------------
__global__ void f32_to_f16_kernel(const float* __restrict__ src,
                                  _Float16* __restrict__ dst, int n) {
  const int i = blockIdx.x*blockDim.x + threadIdx.x;
  if (i < n) dst[i] = (_Float16)src[i];
}

// ---------------- patchify: (B,C,T) -> (B,NP,C*P), dual f32+f16 ----------
__global__ void patchify_kernel(const float* __restrict__ x,
                                float* __restrict__ pF, _Float16* __restrict__ pH) {
  int o = blockIdx.x * blockDim.x + threadIdx.x;
  if (o >= M_*CP_) return;
  int j  = o & 31;
  int c  = (o >> 5) & 15;
  int bp = o >> 9;
  int p  = bp % NP_;
  int b  = bp / NP_;
  const float v = x[((size_t)(b*16 + c))*32000 + p*32 + j];
  pF[o] = v;
  pH[o] = (_Float16)v;
}

// ---- GEMM: out = act(A[M,K](f16) @ W[N,K](f16)^T + bias); 128x64 tile -----
// 256 threads = 8 waves; wave w owns rows w*16..+15 x all 64 cols (4 WMMAs/step).
// Double-buffered async LDS fill: tile k+1 streams in while tile k computes.
#define LDH 40
template<int ACT>
__global__ __launch_bounds__(256)
void gemm_wmma(const _Float16* __restrict__ A, const _Float16* __restrict__ W,
               const float* __restrict__ bias,
               float* __restrict__ outF, _Float16* __restrict__ outH,
               int Ndim, int Kdim) {
  __shared__ _Float16 At[2][128*LDH];
  __shared__ _Float16 Wt[2][64*LDH];
  const int m0 = blockIdx.y * 128;
  const int n0 = blockIdx.x * 64;
  const int t  = threadIdx.x;
  const int w  = t >> 5, lane = t & 31;
  const int wm = w * 16;
  // A: 512 16B-chunks/step -> 2 per thread; W: 256 chunks -> 1 per thread.
  const int ar = t >> 1, ac = (t & 1) * 16;       // A rows 0..127, col 0/16
  const int wr = t >> 2, wc = (t & 3) * 8;        // W rows 0..63
  const _Float16* Arow = &A[(size_t)(m0+ar)*Kdim + ac];
  const _Float16* Wrow = &W[(size_t)(n0+wr)*Kdim + wc];
  v8f acc[4] = {{}, {}, {}, {}};

  // prologue: stage k0 = 0 into buffer 0
  async_copy16(&At[0][ar*LDH + ac],     Arow);
  async_copy16(&At[0][ar*LDH + ac + 8], Arow + 8);
  async_copy16(&Wt[0][wr*LDH + wc],     Wrow);

  int cur = 0;
  for (int k0 = 0; k0 < Kdim; k0 += 32) {
    if (k0 + 32 < Kdim) {      // prefetch next tile into alternate buffer
      const int nxt = cur ^ 1;
      async_copy16(&At[nxt][ar*LDH + ac],     Arow + k0 + 32);
      async_copy16(&At[nxt][ar*LDH + ac + 8], Arow + k0 + 40);
      async_copy16(&Wt[nxt][wr*LDH + wc],     Wrow + k0 + 32);
      wait_async3();           // in-order: current tile's 3 copies done
    } else {
      wait_async0();
    }
    __syncthreads();           // current tile visible to all waves
    v16h af = frag_rowmajor(&At[cur][wm*LDH], LDH);
#pragma unroll
    for (int j = 0; j < 4; ++j) {
      v16h bf = frag_rowmajor(&Wt[cur][(j*16)*LDH], LDH);
      acc[j] = wmma_16x16x32(af, bf, acc[j]);
    }
    __syncthreads();           // all reads done before this buffer is refilled
    cur ^= 1;
  }

  const int nb = lane & 15, half = lane >> 4;
#pragma unroll
  for (int j = 0; j < 4; ++j) {
    const int nn = n0 + j*16 + nb;
    const float bv = bias ? bias[nn] : 0.f;
#pragma unroll
    for (int vr = 0; vr < 8; ++vr) {
      const int m = m0 + wm + vr + half*8;
      float x = acc[j][vr] + bv;
      if (ACT == 1) x = gelu_exact(x);
      if (outF) outF[(size_t)m*Ndim + nn] = x;
      if (outH) outH[(size_t)m*Ndim + nn] = (_Float16)x;
    }
  }
}

// ---------------- fused flash attention (qkv in f16) ----------------------
// grid: B*H*16 blocks of 128 threads (4 waves); wave owns 16 queries.
// 32 key tiles of 32 keys; online softmax; double-buffered async K/V fill.
#define LDK 40
__global__ __launch_bounds__(128)
void attention_wmma(const _Float16* __restrict__ qkv, _Float16* __restrict__ outp) {
  __shared__ _Float16 Kt[2][32*LDK];
  __shared__ _Float16 Vt[2][32*LDK];
  __shared__ _Float16 Qt[4][16*LDK];
  __shared__ float    Sb[4][16*32];
  __shared__ _Float16 Pb[4][16*32];
  __shared__ float    rowM[4][16], rowL[4][16], rowF[4][16];

  const int t = threadIdx.x, w = t >> 5, lane = t & 31;
  const int qblk = blockIdx.x & 15;
  const int bh   = blockIdx.x >> 4;
  const int b = bh >> 3, h = bh & 7;
  const int q0 = qblk*64 + w*16;
  const float scale = 0.17677669529663687f;   // 1/sqrt(DH), applied to scores

  const int kvr = t >> 2, kvc = (t & 3)*8;    // K/V fill pattern
  auto kv_issue = [&](int buf, int kt) {
    const int key = kt*32 + kvr;
    const int kc = key < NP_ ? key : NP_-1;
    const _Float16* kp = &qkv[((size_t)(b*NP_ + kc))*768 + 256 + h*DH_ + kvc];
    async_copy16(&Kt[buf][kvr*LDK + kvc], kp);
    async_copy16(&Vt[buf][kvr*LDK + kvc], kp + 256);
  };

  kv_issue(0, 0);                              // prologue: key tile 0
  for (int idx = lane; idx < 64; idx += 32) {  // Q tile: 64 8-f16 chunks
    int r = idx >> 2, cb = (idx & 3)*8;
    int q = q0 + r; if (q > NP_-1) q = NP_-1;
    *(h8*)&Qt[w][r*LDK + cb] =
        *(const h8*)&qkv[((size_t)(b*NP_ + q))*768 + h*DH_ + cb];
  }
  if (lane < 16) { rowM[w][lane] = -1e30f; rowL[w][lane] = 0.f; }
  __syncthreads();
  v16h qf = frag_rowmajor(&Qt[w][0], LDK);
  v8f o0 = {}; v8f o1 = {};

  int cur = 0;
  for (int kt = 0; kt < 32; ++kt) {
    const int kb = kt*32;
    if (kt < 31) { kv_issue(cur ^ 1, kt + 1); wait_async2(); }
    else         { wait_async0(); }
    __syncthreads();           // current K/V tile visible
    v16h kf0 = frag_rowmajor(&Kt[cur][0], LDK);
    v16h kf1 = frag_rowmajor(&Kt[cur][16*LDK], LDK);
    v8f s0 = {}; v8f s1 = {};
    s0 = wmma_16x16x32(qf, kf0, s0);
    s1 = wmma_16x16x32(qf, kf1, s1);
    {   // spill scaled scores with key masking
      const int n = lane & 15;
      const int key0 = kb + n, key1 = kb + 16 + n;
#pragma unroll
      for (int vr = 0; vr < 8; ++vr) {
        const int r = vr + ((lane >> 4) << 3);
        Sb[w][r*32 + n]      = (key0 < NP_) ? s0[vr]*scale : -1e30f;
        Sb[w][r*32 + 16 + n] = (key1 < NP_) ? s1[vr]*scale : -1e30f;
      }
    }
    __syncthreads();
    if (lane < 16) {            // online softmax row pass
      const int r = lane;
      float mt = -1e30f;
      for (int j = 0; j < 32; ++j) mt = fmaxf(mt, Sb[w][r*32+j]);
      const float mo = rowM[w][r];
      const float mn = fmaxf(mo, mt);
      const float f  = __expf(mo - mn);
      float sum = 0.f;
      for (int j = 0; j < 32; ++j) {
        const float p = __expf(Sb[w][r*32+j] - mn);
        Pb[w][r*32+j] = (_Float16)p;
        sum += p;
      }
      rowL[w][r] = rowL[w][r]*f + sum;
      rowM[w][r] = mn;
      rowF[w][r] = f;
    }
    __syncthreads();
#pragma unroll
    for (int vr = 0; vr < 8; ++vr) {
      const int r = vr + ((lane >> 4) << 3);
      const float f = rowF[w][r];
      o0[vr] *= f; o1[vr] *= f;
    }
    v16h pf  = frag_rowmajor(&Pb[w][0], 32);
    v16h vf0 = frag_colmajor(&Vt[cur][0],  LDK);
    v16h vf1 = frag_colmajor(&Vt[cur][16], LDK);
    o0 = wmma_16x16x32(pf, vf0, o0);
    o1 = wmma_16x16x32(pf, vf1, o1);
    __syncthreads();           // reads done before this buffer is refilled
    cur ^= 1;
  }
  {
    const int n = lane & 15;
#pragma unroll
    for (int vr = 0; vr < 8; ++vr) {
      const int r = vr + ((lane >> 4) << 3);
      const int q = q0 + r;
      if (q < NP_) {
        const float inv = 1.0f / rowL[w][r];
        outp[((size_t)(b*NP_ + q))*D_ + h*DH_ + n]      = (_Float16)(o0[vr]*inv);
        outp[((size_t)(b*NP_ + q))*D_ + h*DH_ + 16 + n] = (_Float16)(o1[vr]*inv);
      }
    }
  }
}

// ---------------- LayerNorm (D=256): f32 in, f32 + f16 out ----------------
__global__ __launch_bounds__(256)
void ln_kernel(const float* __restrict__ in1, const float* __restrict__ in2,
               const float* __restrict__ gamma, const float* __restrict__ beta,
               const float* __restrict__ pos,
               float* __restrict__ outF, _Float16* __restrict__ outH) {
  __shared__ float red[8];
  __shared__ float sM, sV;
  const int row = blockIdx.x;
  const int d = threadIdx.x, w = d >> 5, lane = d & 31;
  float v = in1[(size_t)row*D_ + d];
  if (in2) v += in2[(size_t)row*D_ + d];
  float s = v;
  for (int off = 16; off; off >>= 1) s += __shfl_xor(s, off, 32);
  if (lane == 0) red[w] = s;
  __syncthreads();
  if (d == 0) { float m = 0.f; for (int i = 0; i < 8; ++i) m += red[i]; sM = m / D_; }
  __syncthreads();
  const float m = sM;
  const float dv = v - m;
  float q = dv*dv;
  for (int off = 16; off; off >>= 1) q += __shfl_xor(q, off, 32);
  if (lane == 0) red[w] = q;
  __syncthreads();
  if (d == 0) { float vv = 0.f; for (int i = 0; i < 8; ++i) vv += red[i]; sV = vv / D_; }
  __syncthreads();
  float y = dv * rsqrtf(sV + EPS_) * gamma[d] + beta[d];
  if (pos) y += pos[(size_t)(row % NP_)*D_ + d];
  outF[(size_t)row*D_ + d] = y;
  if (outH) outH[(size_t)row*D_ + d] = (_Float16)y;
}

// ---------------- VQ pieces ----------------
__global__ __launch_bounds__(256)
void cnorm_kernel(const float* __restrict__ cb, float* __restrict__ cn) {
  __shared__ float red[8];
  const int n = blockIdx.x, d = threadIdx.x, w = d >> 5, lane = d & 31;
  float v = cb[(size_t)n*D_ + d]; v *= v;
  for (int off = 16; off; off >>= 1) v += __shfl_xor(v, off, 32);
  if (lane == 0) red[w] = v;
  __syncthreads();
  if (d == 0) { float s = 0.f; for (int i = 0; i < 8; ++i) s += red[i]; cn[n] = s; }
}

__global__ __launch_bounds__(256)
void argmin_kernel(const float* __restrict__ G, const float* __restrict__ cn,
                   int* __restrict__ idx, float* __restrict__ idxf) {
  const int w = threadIdx.x >> 5, lane = threadIdx.x & 31;
  const int row = blockIdx.x*8 + w;
  float best = 1e30f; int bi = 0;
  for (int n = lane; n < K_; n += 32) {
    const float d = cn[n] - 2.f*G[(size_t)row*K_ + n];
    if (d < best || (d == best && n < bi)) { best = d; bi = n; }
  }
  for (int off = 16; off; off >>= 1) {
    const float ov = __shfl_xor(best, off, 32);
    const int   oi = __shfl_xor(bi, off, 32);
    if (ov < best || (ov == best && oi < bi)) { best = ov; bi = oi; }
  }
  if (lane == 0) { idx[row] = bi; idxf[row] = (float)bi; }
}

__global__ void gather_kernel(const float* __restrict__ cb, const int* __restrict__ idx,
                              _Float16* __restrict__ quantH) {
  const int row = blockIdx.x, d = threadIdx.x;
  quantH[(size_t)row*D_ + d] = (_Float16)cb[(size_t)idx[row]*D_ + d];
}

__global__ void copy_kernel(const float* __restrict__ src, float* __restrict__ dst, int n) {
  const int i = blockIdx.x*blockDim.x + threadIdx.x;
  if (i < n) dst[i] = src[i];
}

// ---------------- MSE loss (fixed-order two-pass, deterministic) ----------
__global__ __launch_bounds__(256)
void loss_partial_kernel(const float* __restrict__ recon, const float* __restrict__ patches,
                         float* __restrict__ partial) {
  __shared__ float red[8];
  const int gid = blockIdx.x*256 + threadIdx.x;
  const int stride = gridDim.x*256;
  float s = 0.f;
  for (int i = gid; i < M_*CP_; i += stride) {
    const float d = recon[i] - patches[i];
    s += d*d;
  }
  const int w = threadIdx.x >> 5, lane = threadIdx.x & 31;
  for (int off = 16; off; off >>= 1) s += __shfl_xor(s, off, 32);
  if (lane == 0) red[w] = s;
  __syncthreads();
  if (threadIdx.x == 0) { float t2 = 0.f; for (int i = 0; i < 8; ++i) t2 += red[i]; partial[blockIdx.x] = t2; }
}

__global__ __launch_bounds__(256)
void loss_final_kernel(const float* __restrict__ partial, int n, float* __restrict__ loss) {
  __shared__ float red[8];
  float s = 0.f;
  for (int i = threadIdx.x; i < n; i += 256) s += partial[i];
  const int w = threadIdx.x >> 5, lane = threadIdx.x & 31;
  for (int off = 16; off; off >>= 1) s += __shfl_xor(s, off, 32);
  if (lane == 0) red[w] = s;
  __syncthreads();
  if (threadIdx.x == 0) { float t2 = 0.f; for (int i = 0; i < 8; ++i) t2 += red[i]; loss[0] = t2 / (float)(M_*CP_); }
}

// ---------------- orchestration ----------------
extern "C" void kernel_launch(void* const* d_in, const int* in_sizes, int n_in,
                              void* d_out, int out_size, void* d_ws, size_t ws_size,
                              hipStream_t stream) {
  (void)in_sizes; (void)n_in; (void)out_size; (void)ws_size;
  const float* x    = (const float*)d_in[0];
  const float* pw1  = (const float*)d_in[1];
  const float* pb1  = (const float*)d_in[2];
  const float* pw2  = (const float*)d_in[3];
  const float* pb2  = (const float*)d_in[4];
  const float* plns = (const float*)d_in[5];
  const float* plnb = (const float*)d_in[6];
  const float* pos  = (const float*)d_in[7];
  const float* Wqkv = (const float*)d_in[8];
  const float* bqkv = (const float*)d_in[9];
  const float* Wo   = (const float*)d_in[10];
  const float* bo   = (const float*)d_in[11];
  const float* ln1s = (const float*)d_in[12];
  const float* ln1b = (const float*)d_in[13];
  const float* W1   = (const float*)d_in[14];
  const float* b1   = (const float*)d_in[15];
  const float* W2   = (const float*)d_in[16];
  const float* b2   = (const float*)d_in[17];
  const float* ln2s = (const float*)d_in[18];
  const float* ln2b = (const float*)d_in[19];
  const float* cbk  = (const float*)d_in[20];
  const float* dw1  = (const float*)d_in[21];
  const float* db1  = (const float*)d_in[22];
  const float* dw2  = (const float*)d_in[23];
  const float* db2  = (const float*)d_in[24];

  // ---- f32 workspace (floats from base) ----
  float* ws       = (float*)d_ws;
  float* patchesF = ws;                    //  8,192,000
  float* zF       = ws +  8192000;         //  4,096,000
  float* tmpF     = ws + 12288000;         //  4,096,000 (pe2 / proj / ff2 out)
  float* G        = ws + 16384000;         // 16,384,000 (VQ distances)
  float* cnorm    = ws + 32768000;         //      1,024
  float* partial  = ws + 32769024;         //      4,096
  int*   idxbuf   = (int*)(ws + 32773120); //     16,000
  // ---- f16 arena ----
  _Float16* hb      = (_Float16*)(ws + 32789120);
  _Float16* patchesH= hb;                  //  8,192,000
  _Float16* hidH    = hb +  8192000;       //  8,192,000 (pe hidden / dec hidden)
  _Float16* zH      = hb + 16384000;       //  4,096,000
  _Float16* qkvH    = hb + 20480000;       // 12,288,000
  _Float16* attnH   = hb + 32768000;       //  4,096,000
  _Float16* ffH     = hb + 36864000;       // 16,384,000
  _Float16* quantH  = hb + 53248000;       //  4,096,000
  _Float16* wH      = hb + 57344000;       //  4,194,304 weight shadows
  _Float16* pw1H  = wH;                // 262,144
  _Float16* pw2H  = wH +  262144;      // 131,072
  _Float16* WqkvH = wH +  393216;      // 786,432
  _Float16* WoH   = wH + 1179648;      // 262,144
  _Float16* W1H   = wH + 1441792;      // 1,048,576
  _Float16* W2H   = wH + 2490368;      // 1,048,576
  _Float16* dw1H  = wH + 3538944;      // 131,072
  _Float16* dw2H  = wH + 3670016;      // 262,144
  _Float16* cbkH  = wH + 3932160;      // 262,144

  float* out   = (float*)d_out;            // [recon | loss | indices | z]
  float* recon = out;
  float* lossp = out + 8192000;
  float* idxf  = out + 8192001;
  float* zout  = out + 8208001;

  const dim3 blk256(256), blk128(128);
  auto cvt = [&](const float* s, _Float16* d, int n) {
    f32_to_f16_kernel<<<dim3((n+255)/256), blk256, 0, stream>>>(s, d, n);
  };

  // weight shadows (f16)
  cvt(pw1,  pw1H,  512*512);
  cvt(pw2,  pw2H,  256*512);
  cvt(Wqkv, WqkvH, L_*768*256);
  cvt(Wo,   WoH,   L_*256*256);
  cvt(W1,   W1H,   L_*1024*256);
  cvt(W2,   W2H,   L_*256*1024);
  cvt(dw1,  dw1H,  512*256);
  cvt(dw2,  dw2H,  512*512);
  cvt(cbk,  cbkH,  K_*256);

  // patch embed
  patchify_kernel<<<dim3((M_*CP_+255)/256), blk256, 0, stream>>>(x, patchesF, patchesH);
  gemm_wmma<1><<<dim3(512/64,  M_/128), blk256, 0, stream>>>(patchesH, pw1H, pb1, nullptr, hidH, 512, 512);
  gemm_wmma<0><<<dim3(256/64,  M_/128), blk256, 0, stream>>>(hidH, pw2H, pb2, tmpF, nullptr, 256, 512);
  ln_kernel<<<dim3(M_), blk256, 0, stream>>>(tmpF, nullptr, plns, plnb, pos, zF, zH);

  // transformer
  for (int l = 0; l < L_; ++l) {
    gemm_wmma<0><<<dim3(768/64,  M_/128), blk256, 0, stream>>>(zH, WqkvH + (size_t)l*768*256, bqkv + l*768, nullptr, qkvH, 768, 256);
    attention_wmma<<<dim3(B_*H_*16), blk128, 0, stream>>>(qkvH, attnH);
    gemm_wmma<0><<<dim3(256/64,  M_/128), blk256, 0, stream>>>(attnH, WoH + (size_t)l*256*256, bo + l*256, tmpF, nullptr, 256, 256);
    ln_kernel<<<dim3(M_), blk256, 0, stream>>>(zF, tmpF, ln1s + l*256, ln1b + l*256, nullptr, zF, zH);
    gemm_wmma<1><<<dim3(1024/64, M_/128), blk256, 0, stream>>>(zH, W1H + (size_t)l*1024*256, b1 + l*1024, nullptr, ffH, 1024, 256);
    gemm_wmma<0><<<dim3(256/64,  M_/128), blk256, 0, stream>>>(ffH, W2H + (size_t)l*256*1024, b2 + l*256, tmpF, nullptr, 256, 1024);
    ln_kernel<<<dim3(M_), blk256, 0, stream>>>(zF, tmpF, ln2s + l*256, ln2b + l*256, nullptr, zF, zH);
  }

  // z output + VQ
  copy_kernel<<<dim3((M_*D_+255)/256), blk256, 0, stream>>>(zF, zout, M_*D_);
  cnorm_kernel<<<dim3(K_), blk256, 0, stream>>>(cbk, cnorm);
  gemm_wmma<0><<<dim3(1024/64, M_/128), blk256, 0, stream>>>(zH, cbkH, nullptr, G, nullptr, 1024, 256);
  argmin_kernel<<<dim3(M_/8), blk256, 0, stream>>>(G, cnorm, idxbuf, idxf);
  gather_kernel<<<dim3(M_), blk256, 0, stream>>>(cbk, idxbuf, quantH);

  // decoder + loss
  gemm_wmma<1><<<dim3(512/64, M_/128), blk256, 0, stream>>>(quantH, dw1H, db1, nullptr, hidH, 512, 256);
  gemm_wmma<0><<<dim3(512/64, M_/128), blk256, 0, stream>>>(hidH, dw2H, db2, recon, nullptr, 512, 512);
  loss_partial_kernel<<<dim3(4096), blk256, 0, stream>>>(recon, patchesF, partial);
  loss_final_kernel<<<dim3(1), blk256, 0, stream>>>(partial, 4096, lossp);
}
